// PlmKnrm_23802708755057
// MI455X (gfx1250) — matrix operand order repaired and checked by reading, not compiled, not run on hardware
//
#include <hip/hip_runtime.h>
#include <hip/hip_bf16.h>
#include <stdint.h>

#define B_  256
#define Q_  32
#define DL_ 256
#define D_  768
#define NK  11
#define KC  64      // K-chunk staged in LDS per iteration
#define ROWP 72     // padded LDS row (bf16 elems): 144B stride -> conflict-free b128 reads

typedef __attribute__((ext_vector_type(16))) __bf16 v16bf;
typedef __attribute__((ext_vector_type(8)))  float  v8f;

union Frag16 {
    uint4  q[2];
    v16bf  v;
};

// round-to-nearest-even f32 -> bf16, packed pair
__device__ __forceinline__ uint32_t pack_bf16x2(float lo, float hi) {
    uint32_t a = __float_as_uint(lo);
    uint32_t b = __float_as_uint(hi);
    a = a + 0x7FFFu + ((a >> 16) & 1u);
    b = b + 0x7FFFu + ((b >> 16) & 1u);
    return (a >> 16) | (b & 0xFFFF0000u);
}

__global__ __launch_bounds__(256, 1)
void plm_knrm_kernel(const float* __restrict__ qry,
                     const float* __restrict__ doc,
                     const float* __restrict__ ctx,
                     const float* __restrict__ W,
                     const float* __restrict__ bias,
                     const int*   __restrict__ qlens,
                     const int*   __restrict__ dlens,
                     float*       __restrict__ out)
{
    __shared__ uint16_t q_lds[Q_ * ROWP];       // bf16 query chunk  [32 x 64] padded
    __shared__ uint16_t d_lds[DL_ * ROWP];      // bf16 doc chunk    [256 x 64] padded
    __shared__ float    q_part[Q_][8];          // query sumsq partials
    __shared__ float    inv_qn[Q_];
    __shared__ float    inv_dn[DL_];
    __shared__ float    soft_ws[8][Q_][NK];     // per-wave soft-TF slices (deterministic)
    __shared__ float    log_tf[Q_][NK];
    __shared__ float    wave_sum[8];
    __shared__ float    rbf_contrib[NK];

    const int b    = blockIdx.x;
    const int tid  = threadIdx.x;
    const int lane = tid & 31;
    const int wave = tid >> 5;
    const int kh   = lane >> 4;   // K-half selector for WMMA fragments
    const int nl   = lane & 15;   // row/col-in-tile selector

    v8f zero = {0.f,0.f,0.f,0.f,0.f,0.f,0.f,0.f};
    v8f acc00 = zero, acc01 = zero, acc10 = zero, acc11 = zero;

    // staging assignments (fixed per thread across all K-chunks)
    const float* drow_g = doc + ((size_t)b * DL_ + tid) * D_;    // one doc row / thread
    const int    qr     = tid >> 3;                              // 8 threads / query row
    const int    qc     = (tid & 7) * 8;
    const float* qrow_g = qry + ((size_t)b * Q_ + qr) * D_ + qc;

    float dsq = 0.f, qs = 0.f;

    for (int kc = 0; kc < D_; kc += KC) {
        __syncthreads();   // previous chunk's WMMA reads done

        if (kc + KC < D_) {                       // global_prefetch_b8 next chunk
            __builtin_prefetch(drow_g + kc + KC, 0, 1);
            __builtin_prefetch(qrow_g + kc + KC, 0, 1);
        }

        // ---- stage doc chunk: f32 -> bf16, fused sum-of-squares ----
        {
            const float4* src = reinterpret_cast<const float4*>(drow_g + kc);
            uint32_t*     dst = reinterpret_cast<uint32_t*>(&d_lds[tid * ROWP]);
            #pragma unroll
            for (int j = 0; j < 16; ++j) {
                float4 f = src[j];
                dsq += f.x*f.x + f.y*f.y + f.z*f.z + f.w*f.w;
                dst[2*j]     = pack_bf16x2(f.x, f.y);
                dst[2*j + 1] = pack_bf16x2(f.z, f.w);
            }
        }
        // ---- stage query chunk ----
        {
            const float4* src = reinterpret_cast<const float4*>(qrow_g + kc);
            uint32_t*     dst = reinterpret_cast<uint32_t*>(&q_lds[qr * ROWP + qc]);
            #pragma unroll
            for (int j = 0; j < 2; ++j) {
                float4 f = src[j];
                qs += f.x*f.x + f.y*f.y + f.z*f.z + f.w*f.w;
                dst[2*j]     = pack_bf16x2(f.x, f.y);
                dst[2*j + 1] = pack_bf16x2(f.z, f.w);
            }
        }
        __syncthreads();

        // ---- bf16 WMMA over the chunk: 2 k-steps of 32 ----
        #pragma unroll
        for (int kk = 0; kk < KC; kk += 32) {
            Frag16 a0, a1, b0, b1;
            // A 16x32 bf16 layout: m = lane&15; group0 K = kh*8..+7, group1 K = 16+kh*8..+7
            const uint16_t* qa0 = &q_lds[(nl)      * ROWP + kk + kh*8];
            const uint16_t* qa1 = &q_lds[(16 + nl) * ROWP + kk + kh*8];
            a0.q[0] = *reinterpret_cast<const uint4*>(qa0);
            a0.q[1] = *reinterpret_cast<const uint4*>(qa0 + 16);
            a1.q[0] = *reinterpret_cast<const uint4*>(qa1);
            a1.q[1] = *reinterpret_cast<const uint4*>(qa1 + 16);
            // B 32x16 bf16 layout: n = lane&15; group0 K = kh*8..+7, group1 K = 16+kh*8..+7
            const uint16_t* db0 = &d_lds[(wave*32 + nl)      * ROWP + kk + kh*8];
            const uint16_t* db1 = &d_lds[(wave*32 + 16 + nl) * ROWP + kk + kh*8];
            b0.q[0] = *reinterpret_cast<const uint4*>(db0);
            b0.q[1] = *reinterpret_cast<const uint4*>(db0 + 16);
            b1.q[0] = *reinterpret_cast<const uint4*>(db1);
            b1.q[1] = *reinterpret_cast<const uint4*>(db1 + 16);

            acc00 = __builtin_amdgcn_wmma_f32_16x16x32_bf16(false, a0.v, false, b0.v,
                                                            (short)0, acc00, false, false);
            acc01 = __builtin_amdgcn_wmma_f32_16x16x32_bf16(false, a0.v, false, b1.v,
                                                            (short)0, acc01, false, false);
            acc10 = __builtin_amdgcn_wmma_f32_16x16x32_bf16(false, a1.v, false, b0.v,
                                                            (short)0, acc10, false, false);
            acc11 = __builtin_amdgcn_wmma_f32_16x16x32_bf16(false, a1.v, false, b1.v,
                                                            (short)0, acc11, false, false);
        }
    }

    // ---- norms (no atomics: deterministic) ----
    q_part[qr][tid & 7] = qs;
    inv_dn[tid] = rsqrtf(dsq);
    __syncthreads();
    if (tid < Q_) {
        float s = 0.f;
        #pragma unroll
        for (int j = 0; j < 8; ++j) s += q_part[tid][j];
        inv_qn[tid] = rsqrtf(s);
    }

    // ---- context . W partial (overlapped here) ----
    {
        float cp = 0.f;
        #pragma unroll
        for (int j = 0; j < 3; ++j) {
            int d = tid + j * 256;
            cp += ctx[(size_t)b * D_ + d] * W[NK + d];
        }
        #pragma unroll
        for (int off = 16; off > 0; off >>= 1) cp += __shfl_xor(cp, off, 32);
        if (lane == 0) wave_sum[wave] = cp;
    }
    __syncthreads();

    // ---- epilogue: normalize, mask, RBF, N-reduction ----
    const float means[NK] = {-0.9f,-0.7f,-0.5f,-0.3f,-0.1f,0.1f,0.3f,0.5f,0.7f,0.9f,1.0f};
    const float i2v[NK]   = {50.f,50.f,50.f,50.f,50.f,50.f,50.f,50.f,50.f,50.f,500000.f};
    const int qlen = qlens[b];
    const int dlen = dlens[b];
    const int n0 = wave*32 + nl;
    const int n1 = wave*32 + 16 + nl;
    const float idn0 = inv_dn[n0];
    const float idn1 = inv_dn[n1];
    const bool  d0ok = (n0 <= dlen);
    const bool  d1ok = (n1 <= dlen);

    #pragma unroll
    for (int mt = 0; mt < 2; ++mt) {
        #pragma unroll
        for (int v = 0; v < 8; ++v) {
            // C/D layout: lanes 0-15 hold M=v, lanes 16-31 hold M=v+8; N=lane&15
            int   m  = mt*16 + v + kh*8;
            float iq = inv_qn[m];
            float s0 = (mt == 0 ? acc00[v] : acc10[v]) * iq * idn0;
            float s1 = (mt == 0 ? acc01[v] : acc11[v]) * iq * idn1;
            bool  qok = (m <= qlen);
            s0 = (qok && d0ok) ? s0 : 0.f;   // masked entries still feed soft-TF
            s1 = (qok && d1ok) ? s1 : 0.f;
            #pragma unroll
            for (int k = 0; k < NK; ++k) {
                float e0 = s0 - means[k];
                float e1 = s1 - means[k];
                float r  = __expf(-e0*e0*i2v[k]) + __expf(-e1*e1*i2v[k]);
                r += __shfl_xor(r, 1, 32);
                r += __shfl_xor(r, 2, 32);
                r += __shfl_xor(r, 4, 32);
                r += __shfl_xor(r, 8, 32);   // sum over 16 lanes (distinct n, same m)
                if (nl == 0) soft_ws[wave][m][k] = r;
            }
        }
    }
    __syncthreads();

    // ---- cross-wave reduce (full DL sum), log ----
    for (int idx = tid; idx < Q_ * NK; idx += 256) {
        int m = idx / NK, k = idx - m * NK;
        float s = 0.f;
        #pragma unroll
        for (int w = 0; w < 8; ++w) s += soft_ws[w][m][k];
        log_tf[m][k] = __logf(s);
    }
    __syncthreads();

    if (tid < NK) {
        float f = 0.f;
        #pragma unroll
        for (int m = 0; m < Q_; ++m) f += log_tf[m][tid];
        rbf_contrib[tid] = f * W[tid];
    }
    __syncthreads();

    if (tid == 0) {
        float tot = bias[0];
        #pragma unroll
        for (int k = 0; k < NK; ++k) tot += rbf_contrib[k];
        #pragma unroll
        for (int w = 0; w < 8; ++w) tot += wave_sum[w];
        out[b] = tanhf(tot);
    }
}

extern "C" void kernel_launch(void* const* d_in, const int* in_sizes, int n_in,
                              void* d_out, int out_size, void* d_ws, size_t ws_size,
                              hipStream_t stream) {
    const float* qry  = (const float*)d_in[0];   // [B,Q,D]
    const float* doc  = (const float*)d_in[1];   // [B,DL,D]
    const float* ctx  = (const float*)d_in[2];   // [B,D]
    const float* W    = (const float*)d_in[3];   // [NK+D,1]
    const float* bias = (const float*)d_in[4];   // [1]
    const int*   ql   = (const int*)d_in[5];     // [B]
    const int*   dl   = (const int*)d_in[6];     // [B]
    float* out = (float*)d_out;                  // [B,1]

    plm_knrm_kernel<<<dim3(B_), dim3(256), 0, stream>>>(qry, doc, ctx, W, bias, ql, dl, out);
}
